// MultiHeadAttetion_23313082482695
// MI455X (gfx1250) — matrix operand order repaired and checked by reading, not compiled, not run on hardware
//
#include <hip/hip_runtime.h>
#include <math.h>

typedef __bf16 bf16_t;
typedef __attribute__((ext_vector_type(16))) __bf16 v16bf;
typedef __attribute__((ext_vector_type(8)))  __bf16 v8bf;
typedef __attribute__((ext_vector_type(8)))  float  v8f;

// Problem constants (fixed by the reference)
constexpr int Bc  = 2;
constexpr int Sc  = 2048;
constexpr int Dc  = 1024;
constexpr int Hc  = 16;
constexpr int HDc = 64;
constexpr int Mc  = Bc * Sc;   // 4096
constexpr int Kc  = Dc;        // 1024
constexpr int Nc  = Dc;        // 1024

static __device__ __forceinline__ v8f wmma_bf16(const v16bf& a, const v16bf& b, const v8f& c) {
  return __builtin_amdgcn_wmma_f32_16x16x32_bf16(false, a, false, b, (short)0, c, false, false);
}

static __device__ __forceinline__ v16bf load_v16bf(const bf16_t* p0, const bf16_t* p1) {
  v8bf lo = *(const v8bf*)p0;
  v8bf hi = *(const v8bf*)p1;
  return __builtin_shufflevector(lo, hi, 0,1,2,3,4,5,6,7,8,9,10,11,12,13,14,15);
}

// Async global -> LDS copy of 16 bytes (tracked by ASYNCcnt), GV addressing.
static __device__ __forceinline__ void async_copy_b128(void* lds_ptr, const void* gptr) {
  unsigned lds = (unsigned)(uintptr_t)lds_ptr;          // low 32 bits = LDS byte addr
  unsigned long long ga = (unsigned long long)(uintptr_t)gptr;
  asm volatile("global_load_async_to_lds_b128 %0, %1, off"
               :: "v"(lds), "v"(ga) : "memory");
}

// ---------------------------------------------------------------------------
// fp32 -> bf16 conversion, 8 elements/thread (n must be a multiple of 8)
// ---------------------------------------------------------------------------
__global__ void convert_f32_bf16(const float* __restrict__ src,
                                 bf16_t* __restrict__ dst, int n) {
  int i = (blockIdx.x * blockDim.x + threadIdx.x) * 8;
  if (i < n) {
    float4 a = *(const float4*)(src + i);
    float4 b = *(const float4*)(src + i + 4);
    v8bf o;
    o[0] = (bf16_t)a.x; o[1] = (bf16_t)a.y; o[2] = (bf16_t)a.z; o[3] = (bf16_t)a.w;
    o[4] = (bf16_t)b.x; o[5] = (bf16_t)b.y; o[6] = (bf16_t)b.z; o[7] = (bf16_t)b.w;
    *(v8bf*)(dst + i) = o;
  }
}

// ---------------------------------------------------------------------------
// GEMM: C[m,n] = (sum_k A[m,k] * W[n,k] + bias[n]) * scale   (all bf16 in, f32 acc)
// Double-buffered LDS tiles filled with GLOBAL_LOAD_ASYNC_TO_LDS_B128.
// mode 0: fp32 out row-major [M,N]
// mode 1: bf16 out heads layout [B,H,S,HD]
// mode 2: bf16 out transposed heads layout [B,H,HD,S]
// Tile: 128(M) x 64(N), 256 threads = 8 waves (wave32), BK = 32.
// ---------------------------------------------------------------------------
__global__ __launch_bounds__(256)
void gemm_bias_wmma(const bf16_t* __restrict__ Ab, const bf16_t* __restrict__ Wb,
                    const float* __restrict__ bias,
                    float* __restrict__ outF, bf16_t* __restrict__ outBf,
                    float scale, int mode) {
  constexpr int BM = 128, BN = 64, BK = 32;
  __shared__ bf16_t As[2][BM][BK];
  __shared__ bf16_t Bs[2][BN][BK];

  const int tid  = threadIdx.x;
  const int lane = tid & 31;
  const int wave = tid >> 5;           // 0..7
  const int wm   = wave >> 1;          // 0..3 (M, 32 rows each)
  const int wn   = wave & 1;           // 0..1 (N, 32 cols each)
  const int half = lane >> 4;          // 0 or 1
  const int lrow = lane & 15;
  const int kh   = half * 8;           // unused; A fragments also use k0l
  const int k0l  = half * 16;          // B-fragment K sub-offset

  const int m0 = blockIdx.y * BM;
  const int n0 = blockIdx.x * BN;

  // per-thread staging coordinates
  const int ar = tid >> 1, ac = (tid & 1) * 16;   // A: 2 x b128 per thread
  const int br = tid >> 2, bc = (tid & 3) * 8;    // B: 1 x b128 per thread

  v8f acc[2][2];
#pragma unroll
  for (int i = 0; i < 2; ++i)
#pragma unroll
    for (int j = 0; j < 2; ++j)
      acc[i][j] = v8f{0.f,0.f,0.f,0.f,0.f,0.f,0.f,0.f};

  auto issue_tile = [&](int kt, int buf) {
    const bf16_t* ag = Ab + (size_t)(m0 + ar) * Kc + kt * BK + ac;
    async_copy_b128(&As[buf][ar][ac],     ag);
    async_copy_b128(&As[buf][ar][ac + 8], ag + 8);
    const bf16_t* bg = Wb + (size_t)(n0 + br) * Kc + kt * BK + bc;
    async_copy_b128(&Bs[buf][br][bc], bg);
  };

  constexpr int NT = Kc / BK;  // 32 K-tiles
  issue_tile(0, 0);

  for (int kt = 0; kt < NT; ++kt) {
    const int cur = kt & 1;
    if (kt + 1 < NT) {
      issue_tile(kt + 1, cur ^ 1);
      asm volatile("s_wait_asynccnt 0x3" ::: "memory");  // oldest 3 (current tile) done
    } else {
      asm volatile("s_wait_asynccnt 0x0" ::: "memory");
    }
    __syncthreads();

    v16bf afr[2], bfr[2];
#pragma unroll
    for (int tm = 0; tm < 2; ++tm) {
      const bf16_t* rp = &As[cur][wm * 32 + tm * 16 + lrow][0];
      afr[tm] = load_v16bf(rp + k0l, rp + k0l + 8);
    }
#pragma unroll
    for (int tn = 0; tn < 2; ++tn) {
      const bf16_t* cp = &Bs[cur][wn * 32 + tn * 16 + lrow][0] + k0l;
      bfr[tn] = load_v16bf(cp, cp + 8);
    }
#pragma unroll
    for (int tm = 0; tm < 2; ++tm)
#pragma unroll
      for (int tn = 0; tn < 2; ++tn)
        acc[tm][tn] = wmma_bf16(afr[tm], bfr[tn], acc[tm][tn]);
    __syncthreads();   // all waves done with buf `cur` before it is refilled
  }

  // --- epilogue: bias + scale, scatter per mode ---
#pragma unroll
  for (int tm = 0; tm < 2; ++tm) {
#pragma unroll
    for (int tn = 0; tn < 2; ++tn) {
      const int n = n0 + wn * 32 + tn * 16 + lrow;
      const float bsc = bias[n];
#pragma unroll
      for (int r = 0; r < 8; ++r) {
        const int m = m0 + wm * 32 + tm * 16 + half * 8 + r;
        const float val = (acc[tm][tn][r] + bsc) * scale;
        if (mode == 0) {
          outF[(size_t)m * Nc + n] = val;
        } else {
          const int b = m >> 11, s = m & (Sc - 1);
          const int h = n >> 6,  hd = n & (HDc - 1);
          size_t idx = (mode == 1)
              ? ((((size_t)b * Hc + h) * Sc + s) * HDc + hd)
              : ((((size_t)b * Hc + h) * HDc + hd) * Sc + s);
          outBf[idx] = (bf16_t)val;
        }
      }
    }
  }
}

// ---------------------------------------------------------------------------
// Fused attention: one block = one (b,h) and one 16-row q block.
//   scores (16 x S, fp32) live in LDS; softmax in fp32; single non-temporal
//   write of atten; P@V from LDS with on-the-fly bf16 conversion.
// Block: 128 threads (4 waves). Dynamic LDS: 16*S*4 + reduction buffers.
// ---------------------------------------------------------------------------
__global__ __launch_bounds__(128)
void attn_fused(const bf16_t* __restrict__ Qh,   // [B,H,S,HD] (pre-scaled)
                const bf16_t* __restrict__ Kh,   // [B,H,S,HD]
                const bf16_t* __restrict__ VhT,  // [B,H,HD,S]
                bf16_t* __restrict__ Ctx,        // [B,S,D] merged heads
                float* __restrict__ Atten) {     // [B,H,S,S]
  extern __shared__ char smem[];
  float* sc     = (float*)smem;                             // 16 * S
  float* redmx  = (float*)(smem + (size_t)16 * Sc * 4);     // 16*8
  float* redsum = redmx + 16 * 8;                           // 16*8
  float* rowinv = redsum + 16 * 8;                          // 16

  const int tid  = threadIdx.x;
  const int lane = tid & 31;
  const int wave = tid >> 5;          // 0..3
  const int half = lane >> 4;
  const int lrow = lane & 15;
  const int kh   = half * 8;
  const int k0l  = half * 16;

  const int bh   = blockIdx.x / (Sc / 16);
  const int qblk = blockIdx.x % (Sc / 16);
  const int q0   = qblk * 16;
  const int b    = bh / Hc;
  const int h    = bh % Hc;

  // ---- Q fragments (16 x 64 = two K=32 fragments), per-wave copy ----
  const bf16_t* qrow = Qh + ((size_t)bh * Sc + q0 + lrow) * HDc;
  const v16bf aq0 = load_v16bf(qrow + k0l,      qrow + k0l + 8);
  const v16bf aq1 = load_v16bf(qrow + 32 + k0l, qrow + 32 + k0l + 8);

  // ---- scores: each wave owns a 512-column strip ----
  const bf16_t* kbase = Kh + (size_t)bh * Sc * HDc;
  for (int ct = wave * 512; ct < wave * 512 + 512; ct += 16) {
    const bf16_t* krow = kbase + (size_t)(ct + lrow) * HDc;
    const v16bf bk0 = load_v16bf(krow + k0l,      krow + k0l + 8);
    const v16bf bk1 = load_v16bf(krow + 32 + k0l, krow + 32 + k0l + 8);
    v8f acc = v8f{0.f,0.f,0.f,0.f,0.f,0.f,0.f,0.f};
    acc = wmma_bf16(aq0, bk0, acc);
    acc = wmma_bf16(aq1, bk1, acc);
    const int mrow = half * 8;
#pragma unroll
    for (int r = 0; r < 8; ++r)
      sc[(size_t)(mrow + r) * Sc + ct + lrow] = acc[r];
  }
  __syncthreads();

  // ---- softmax over 2048 columns; 8 threads per row ----
  const int row = tid >> 3;
  const int seg = tid & 7;
  const int c0  = seg * 256;
  float* srow = sc + (size_t)row * Sc;

  float mx = -INFINITY;
  for (int c = c0; c < c0 + 256; ++c) mx = fmaxf(mx, srow[c]);
  redmx[row * 8 + seg] = mx;
  __syncthreads();

  float rmax = redmx[row * 8];
#pragma unroll
  for (int i = 1; i < 8; ++i) rmax = fmaxf(rmax, redmx[row * 8 + i]);

  float sum = 0.f;
  for (int c = c0; c < c0 + 256; ++c) {
    float p = __expf(srow[c] - rmax);
    srow[c] = p;
    sum += p;
  }
  redsum[row * 8 + seg] = sum;
  __syncthreads();

  float rsum = 0.f;
#pragma unroll
  for (int i = 0; i < 8; ++i) rsum += redsum[row * 8 + i];
  if (seg == 0) rowinv[row] = 1.0f / rsum;
  __syncthreads();

  // ---- normalize + single coalesced non-temporal write of atten ----
  float* abase = Atten + ((size_t)bh * Sc + q0) * Sc;
  for (int idx = tid; idx < 16 * Sc; idx += 128) {
    float p = sc[idx] * rowinv[idx >> 11];
    sc[idx] = p;
    __builtin_nontemporal_store(p, abase + idx);
  }
  __syncthreads();

  // ---- ctx = P @ V : each wave computes a 16x16 slab of the 16x64 result ----
  const int n0 = wave * 16;
  const int hd = n0 + lrow;
  const bf16_t* vcol = VhT + ((size_t)bh * HDc + hd) * Sc;
  v8f cacc = v8f{0.f,0.f,0.f,0.f,0.f,0.f,0.f,0.f};
  const float* prow = sc + (size_t)lrow * Sc;
  for (int k = 0; k < Sc; k += 32) {
    v16bf pa;
#pragma unroll
    for (int i = 0; i < 8; ++i) {
      pa[i]     = (bf16_t)prow[k + k0l + i];
      pa[8 + i] = (bf16_t)prow[k + k0l + 8 + i];
    }
    const v16bf pb = load_v16bf(vcol + k + k0l, vcol + k + k0l + 8);
    cacc = wmma_bf16(pa, pb, cacc);
  }
  const int mrow = half * 8;
#pragma unroll
  for (int r = 0; r < 8; ++r)
    Ctx[((size_t)b * Sc + q0 + mrow + r) * Dc + h * HDc + hd] = (bf16_t)cacc[r];
}

// ---------------------------------------------------------------------------
extern "C" void kernel_launch(void* const* d_in, const int* in_sizes, int n_in,
                              void* d_out, int out_size, void* d_ws, size_t ws_size,
                              hipStream_t stream) {
  const float* q  = (const float*)d_in[0];
  const float* k  = (const float*)d_in[1];
  const float* v  = (const float*)d_in[2];
  const float* Wq = (const float*)d_in[3];
  const float* bq = (const float*)d_in[4];
  const float* Wk = (const float*)d_in[5];
  const float* bk = (const float*)d_in[6];
  const float* Wv = (const float*)d_in[7];
  const float* bv = (const float*)d_in[8];
  const float* Wo = (const float*)d_in[9];
  const float* bo = (const float*)d_in[10];

  float* outp  = (float*)d_out;                         // [B,S,D] fp32
  float* atten = outp + (size_t)Bc * Sc * Dc;           // [B,H,S,S] fp32

  // workspace layout (bf16)
  constexpr size_t WELEM = (size_t)Dc * Dc;             // 1M
  constexpr size_t AELEM = (size_t)Bc * Sc * Dc;        // 4M
  bf16_t* Wqb = (bf16_t*)d_ws;
  bf16_t* Wkb = Wqb + WELEM;
  bf16_t* Wvb = Wkb + WELEM;
  bf16_t* Wob = Wvb + WELEM;
  bf16_t* qb  = Wob + WELEM;
  bf16_t* kb  = qb  + AELEM;
  bf16_t* vb  = kb  + AELEM;
  bf16_t* QhB = vb  + AELEM;
  bf16_t* KhB = QhB + AELEM;
  bf16_t* VtB = KhB + AELEM;
  bf16_t* CtxB = VtB + AELEM;

  // 1) fp32 -> bf16 (weights + activations)
  {
    dim3 blk(256);
    dim3 gW((unsigned)(WELEM / (256 * 8)));
    convert_f32_bf16<<<gW, blk, 0, stream>>>(Wq, Wqb, (int)WELEM);
    convert_f32_bf16<<<gW, blk, 0, stream>>>(Wk, Wkb, (int)WELEM);
    convert_f32_bf16<<<gW, blk, 0, stream>>>(Wv, Wvb, (int)WELEM);
    convert_f32_bf16<<<gW, blk, 0, stream>>>(Wo, Wob, (int)WELEM);
    dim3 gA((unsigned)(AELEM / (256 * 8)));
    convert_f32_bf16<<<gA, blk, 0, stream>>>(q, qb, (int)AELEM);
    convert_f32_bf16<<<gA, blk, 0, stream>>>(k, kb, (int)AELEM);
    convert_f32_bf16<<<gA, blk, 0, stream>>>(v, vb, (int)AELEM);
  }

  // 2) projections (Q pre-scaled by 1/sqrt(HD) = 0.125)
  {
    dim3 blk(256), grd(Nc / 64, Mc / 128);
    gemm_bias_wmma<<<grd, blk, 0, stream>>>(qb, Wqb, bq, nullptr, QhB, 0.125f, 1);
    gemm_bias_wmma<<<grd, blk, 0, stream>>>(kb, Wkb, bk, nullptr, KhB, 1.0f,   1);
    gemm_bias_wmma<<<grd, blk, 0, stream>>>(vb, Wvb, bv, nullptr, VtB, 1.0f,   2);
  }

  // 3) fused attention (scores -> softmax -> atten write -> P@V)
  {
    const size_t shmem = (size_t)16 * Sc * 4 + 2 * 16 * 8 * 4 + 16 * 4;
    dim3 blk(128), grd(Bc * Hc * (Sc / 16));
    attn_fused<<<grd, blk, shmem, stream>>>(QhB, KhB, VtB, CtxB, atten);
  }

  // 4) output projection
  {
    dim3 blk(256), grd(Nc / 64, Mc / 128);
    gemm_bias_wmma<<<grd, blk, 0, stream>>>(CtxB, Wob, bo, outp, nullptr, 1.0f, 0);
  }
}
